// RGCNNet_28991029248370
// MI455X (gfx1250) — matrix operand order repaired and checked by reading, not compiled, not run on hardware
//
#include <hip/hip_runtime.h>
#include <math.h>

// ---------------------------------------------------------------------------
// RGCN QA-GNN forward for MI455X (gfx1250, wave32, WMMA).
// Heavy GEMMs use v_wmma_f32_16x16x32_bf16 (bf16 in, f32 accumulate).
// All fragment loads are UNPREDICATED (clamped addresses); out-of-range
// data only ever reaches accumulator elements that are never stored.
// ---------------------------------------------------------------------------

typedef __bf16 bf16;
typedef __attribute__((ext_vector_type(2)))  __bf16 v2bf;
typedef __attribute__((ext_vector_type(16))) __bf16 v16bf;
typedef __attribute__((ext_vector_type(8)))  float  v8f;

#define B_    32
#define NN    200          // nodes
#define NREL  34
#define CIN   1024
#define CD    200          // concept dim
#define SD    1024
#define FCD   512
#define NLAY  3            // num_gnn_layers + 1
#define NP    224          // padded node & dim (7*32, 14*16)
#define WCOLS 6800         // CD * NREL

__device__ __forceinline__ float gelu_erf(float x) {
  return 0.5f * x * (1.0f + erff(x * 0.70710678118654752f));
}

// K offsets (pair base) for the 16x32 bf16 A fragment (ISA 7.12.2):
// lanes 0-15: VGPR0..3 -> K={0,1},{2,3},{4,5},{6,7}; VGPR4..7 -> K=16..23
// lanes 16-31: +8 within each half.
__device__ __forceinline__ void a_frag_kidx(int lane, int* ka) {
  int kb = ((lane >> 4) & 1) * 8;
#pragma unroll
  for (int g = 0; g < 8; ++g)
    ka[g] = (g < 4) ? (kb + 2 * g) : (16 + kb + 2 * (g - 4));
}

// ---------------------------------------------------------------------------
// rdeg[b,r,i] = 1 / max(sum_j adj[b,r,j,i], 1)
// ---------------------------------------------------------------------------
__global__ void k_rdeg(const float* __restrict__ adj, float* __restrict__ rdeg) {
  int tid = blockIdx.x * blockDim.x + threadIdx.x;
  if (tid >= B_ * NREL * NN) return;
  int i = tid % NN;
  int br = tid / NN;
  const float* base = adj + (size_t)br * NN * NN + i;
  float s = 0.f;
  for (int j = 0; j < NN; ++j) s += base[(size_t)j * NN];
  rdeg[tid] = 1.0f / fmaxf(s, 1.0f);
}

// ---------------------------------------------------------------------------
// xb[b, m<200, n<200] = GELU(emb[concepts[b,m]] @ cptW + cptb)   (bf16, padded)
// One wave per 16x16 tile; WMMA over K=1024 (32 steps). No load predication:
// row/col indices are clamped; garbage lands only in unstored C elements.
// ---------------------------------------------------------------------------
__global__ void __launch_bounds__(128)
k_embed_gemm(const int* __restrict__ concepts, const float* __restrict__ emb,
             const float* __restrict__ cptW, const float* __restrict__ cptb,
             bf16* __restrict__ xb) {
  const int wid = threadIdx.x >> 5, lane = threadIdx.x & 31;
  const int MT = 13, NT = 13;                      // ceil(200/16)
  int tile = blockIdx.x * 4 + wid;
  if (tile >= B_ * MT * NT) return;                // uniform per wave
  int b   = tile / (MT * NT);
  int rem = tile % (MT * NT);
  int m0  = (rem / NT) * 16;
  int n0  = (rem % NT) * 16;

  int ka[8]; a_frag_kidx(lane, ka);
  const int m  = lane & 15;
  const int nl = lane & 15;
  const int kb2 = ((lane >> 4) & 1) * 8;
  const int mrow = ((m0 + m)  < NN) ? (m0 + m)  : 0;        // clamped
  const int gn   = ((n0 + nl) < CD) ? (n0 + nl) : (CD - 1); // clamped
  const float* rowp = emb + (size_t)concepts[b * NN + mrow] * CIN;

  v8f acc = {0.f, 0.f, 0.f, 0.f, 0.f, 0.f, 0.f, 0.f};
  for (int k0 = 0; k0 < CIN; k0 += 32) {
    v16bf a, bm;
#pragma unroll
    for (int g = 0; g < 8; ++g) {
      float2 p = *(const float2*)(rowp + k0 + ka[g]);       // 8-byte load
      a[2 * g]     = (bf16)p.x;
      a[2 * g + 1] = (bf16)p.y;
    }
#pragma unroll
    for (int e = 0; e < 16; ++e)
      bm[e] = (bf16)cptW[(size_t)(k0 + kb2 + e + (e & 8)) * CD + gn];
    acc = __builtin_amdgcn_wmma_f32_16x16x32_bf16(false, a, false, bm,
                                                  (short)0, acc, false, false);
  }
  const int mb = ((lane >> 4) & 1) * 8;
  const float bias = cptb[gn];
#pragma unroll
  for (int d = 0; d < 8; ++d) {
    int gm = m0 + mb + d, go = n0 + nl;
    if (gm < NN && go < CD) {
      float v = gelu_erf(acc[d] + bias);
      xb[((size_t)b * NP + gm) * NP + go] = (bf16)v;
    }
  }
}

// ---------------------------------------------------------------------------
// wb[k<224, n] = bf16(gnn_w[l][k, n]) (zero pad rows 200..223)
// ---------------------------------------------------------------------------
__global__ void k_convert_w(const float* __restrict__ gw, bf16* __restrict__ wb) {
  int tid = blockIdx.x * blockDim.x + threadIdx.x;
  if (tid >= NP * WCOLS) return;
  int k = tid / WCOLS, n = tid % WCOLS;
  float v = (k < CD) ? gw[(size_t)k * WCOLS + n] : 0.f;
  wb[tid] = (bf16)v;
}

// ---------------------------------------------------------------------------
// h[b, j<224, c<6800] = xb[b] @ wb   (bf16 out, f32 accumulate via WMMA)
// Fully padded operands: zero guards, straight-line loads.
// ---------------------------------------------------------------------------
__global__ void __launch_bounds__(128)
k_layer_gemm(const bf16* __restrict__ xb, const bf16* __restrict__ wb,
             bf16* __restrict__ h) {
  const int wid = threadIdx.x >> 5, lane = threadIdx.x & 31;
  const int MT = NP / 16, NT = WCOLS / 16;         // 14, 425
  int tile = blockIdx.x * 4 + wid;
  if (tile >= B_ * MT * NT) return;
  int b   = tile / (MT * NT);
  int rem = tile % (MT * NT);
  int m0  = (rem / NT) * 16;
  int n0  = (rem % NT) * 16;

  int ka[8]; a_frag_kidx(lane, ka);
  const bf16* arow = xb + ((size_t)b * NP + m0 + (lane & 15)) * NP;
  const int nl  = n0 + (lane & 15);
  const int kb2 = ((lane >> 4) & 1) * 8;

  v8f acc = {0.f, 0.f, 0.f, 0.f, 0.f, 0.f, 0.f, 0.f};
  for (int k0 = 0; k0 < NP; k0 += 32) {            // 7 steps (zero-padded K)
    v16bf a, bm;
#pragma unroll
    for (int g = 0; g < 8; ++g) {
      v2bf p = *(const v2bf*)(arow + k0 + ka[g]);  // 4-byte load (pairs)
      a[2 * g]     = p.x;
      a[2 * g + 1] = p.y;
    }
#pragma unroll
    for (int e = 0; e < 16; ++e)
      bm[e] = wb[(size_t)(k0 + kb2 + e + (e & 8)) * WCOLS + nl];
    acc = __builtin_amdgcn_wmma_f32_16x16x32_bf16(false, a, false, bm,
                                                  (short)0, acc, false, false);
  }
  const int mb = ((lane >> 4) & 1) * 8;
#pragma unroll
  for (int d = 0; d < 8; ++d)
    h[((size_t)b * NP + m0 + mb + d) * WCOLS + nl] = (bf16)acc[d];
}

// ---------------------------------------------------------------------------
// out[b,i,o] = GELU( sum_r sum_j (adj[b,r,j,i]*rdeg[b,r,i]) * h[b,j,o*34+r] )
// Block = 4 waves sharing one (b, i-tile); A tile (transposed+normalized)
// staged once in LDS per k-step and reused by all 4 o-tiles. A garbage at
// j>=200 is harmless: h rows j>=200 are exact zeros. Prefetch next adj slice.
// ---------------------------------------------------------------------------
__global__ void __launch_bounds__(128)
k_agg_gemm(const float* __restrict__ adj, const float* __restrict__ rdeg,
           const bf16* __restrict__ h, bf16* __restrict__ xbn,
           float* __restrict__ xf) {
  __shared__ bf16 ldsA[16 * 34];                   // [i][j], padded stride
  const int tid = threadIdx.x;
  const int wid = tid >> 5, lane = tid & 31;
  const int b  = blockIdx.z;
  const int i0 = blockIdx.y * 16;                  // 13 i-tiles
  const int ot = blockIdx.x * 4 + wid;             // o-tile (valid < 13)
  const int o0 = ot * 16;

  int ka[8]; a_frag_kidx(lane, ka);
  const int nl   = lane & 15;
  const int kb2  = ((lane >> 4) & 1) * 8;
  const int mloc = lane & 15;
  const int oc   = ((o0 + nl) < CD) ? (o0 + nl) : (CD - 1);  // clamped col
  const size_t cbase = (size_t)oc * NREL;                    // c = o*34 + r

  v8f acc = {0.f, 0.f, 0.f, 0.f, 0.f, 0.f, 0.f, 0.f};
  for (int r = 0; r < NREL; ++r) {
    const float* adjbr = adj  + ((size_t)b * NREL + r) * NN * NN;
    const float* rdbr  = rdeg + ((size_t)b * NREL + r) * NN;
    for (int j0 = 0; j0 < NP; j0 += 32) {          // K over nodes, 7 steps
      // stage A[i][j] = adj[b,r,j0+j,i0+i] * rdeg[b,r,i0+i]  (coalesced in i)
#pragma unroll
      for (int u = 0; u < 4; ++u) {
        int idx = u * 128 + tid;
        int j = idx >> 4, i = idx & 15;
        int gi = i0 + i, gj = j0 + j;
        int cgi = (gi < NN) ? gi : (NN - 1);       // clamped addresses,
        int cgj = (gj < NN) ? gj : (NN - 1);       // unpredicated loads
        ldsA[i * 34 + j] = (bf16)(adjbr[(size_t)cgj * NN + cgi] * rdbr[cgi]);
      }
      // prefetch next adjacency k-slice (global_prefetch path)
      {
        int pj = j0 + 32 + (tid >> 4);
        int cgi = (i0 + (tid & 15) < NN) ? (i0 + (tid & 15)) : (NN - 1);
        if (pj < NN)
          __builtin_prefetch(adjbr + (size_t)pj * NN + cgi, 0, 1);
      }
      __syncthreads();
      v16bf a, bm;
#pragma unroll
      for (int g = 0; g < 8; ++g) {
        a[2 * g]     = ldsA[mloc * 34 + ka[g]];
        a[2 * g + 1] = ldsA[mloc * 34 + ka[g] + 1];
      }
#pragma unroll
      for (int e = 0; e < 16; ++e)
        bm[e] = h[((size_t)b * NP + j0 + kb2 + e + (e & 8)) * WCOLS + cbase + r];
      acc = __builtin_amdgcn_wmma_f32_16x16x32_bf16(false, a, false, bm,
                                                    (short)0, acc, false, false);
      __syncthreads();
    }
  }
  const int mb = ((lane >> 4) & 1) * 8;
#pragma unroll
  for (int d = 0; d < 8; ++d) {
    int gi = i0 + mb + d, go = o0 + nl;
    if (gi < NN && go < CD) {
      float v = gelu_erf(acc[d]);
      xbn[((size_t)b * NP + gi) * NP + go] = (bf16)v;
      xf[((size_t)b * NN + gi) * CD + go]  = v;
    }
  }
}

// ---------------------------------------------------------------------------
// Small epilogue kernels (attention pooling + MLP head), plain f32 VALU.
// ---------------------------------------------------------------------------
__global__ void k_qs(const float* __restrict__ sent, const float* __restrict__ wq,
                     const float* __restrict__ bq, float* __restrict__ qs) {
  int tid = blockIdx.x * blockDim.x + threadIdx.x;
  if (tid >= B_ * CD) return;
  int b = tid / CD, j = tid % CD;
  float s = bq[j];
  for (int k = 0; k < SD; ++k) s += sent[b * SD + k] * wq[(size_t)k * CD + j];
  qs[tid] = s;
}

__global__ void k_kv(const float* __restrict__ xf, const float* __restrict__ wk,
                     const float* __restrict__ bk, const float* __restrict__ wv,
                     const float* __restrict__ bv, float* __restrict__ ksb,
                     float* __restrict__ vsb) {
  int tid = blockIdx.x * blockDim.x + threadIdx.x;
  if (tid >= B_ * NN * CD) return;
  int j = tid % CD, bl = tid / CD;
  float sk = bk[j], sv = bv[j];
  const float* xr = xf + (size_t)bl * CD;
  for (int k = 0; k < CD; ++k) {
    float x = xr[k];
    sk += x * wk[(size_t)k * CD + j];
    sv += x * wv[(size_t)k * CD + j];
  }
  ksb[tid] = sk;
  vsb[tid] = sv;
}

__global__ void k_attn(const float* __restrict__ qs, const float* __restrict__ ksb,
                       float* __restrict__ attnw) {
  int tid = blockIdx.x * blockDim.x + threadIdx.x;
  if (tid >= B_ * 2 * NN) return;
  int l = tid % NN, bh = tid / NN, b = bh >> 1, hd = bh & 1;
  const float* q  = qs  + b * CD + hd * 100;
  const float* kk = ksb + ((size_t)(b * NN + l)) * CD + hd * 100;
  float s = 0.f;
  for (int d = 0; d < 100; ++d) s += q[d] * kk[d];
  attnw[tid] = s * 0.1f;                            // 1/sqrt(dk=100)
}

__global__ void k_softmax(const int* __restrict__ adj_len, float* __restrict__ attnw,
                          float* __restrict__ out_pool) {
  int tid = blockIdx.x * blockDim.x + threadIdx.x;
  if (tid >= B_ * 2) return;
  int b = tid >> 1, hd = tid & 1;
  int len = adj_len[b];
  len = (len < 1) ? 1 : (len > NN ? NN : len);
  float* row = attnw + (size_t)tid * NN;
  float mx = -1e30f;
  for (int l = 0; l < len; ++l) mx = fmaxf(mx, row[l]);
  float den = 0.f;
  for (int l = 0; l < len; ++l) den += expf(row[l] - mx);
  float inv = 1.0f / den;
  float* op = out_pool + ((size_t)(hd * B_ + b)) * NN;   // (H,B,L) layout
  for (int l = 0; l < NN; ++l) {
    float p = (l < len) ? expf(row[l] - mx) * inv : 0.f;
    row[l] = p;
    op[l]  = p;
  }
}

__global__ void k_pool(const float* __restrict__ attnw, const float* __restrict__ vsb,
                       float* __restrict__ gv) {
  int tid = blockIdx.x * blockDim.x + threadIdx.x;
  if (tid >= B_ * CD) return;
  int b = tid / CD, j = tid % CD, hd = j / 100;
  const float* arow = attnw + (size_t)(b * 2 + hd) * NN;
  float s = 0.f;
  for (int l = 0; l < NN; ++l)
    s += arow[l] * vsb[((size_t)(b * NN + l)) * CD + j];
  gv[tid] = s;
}

__global__ void k_mlp1(const float* __restrict__ gv, const float* __restrict__ sent,
                       const float* __restrict__ W0, const float* __restrict__ b0,
                       float* __restrict__ h1) {
  int tid = blockIdx.x * blockDim.x + threadIdx.x;
  if (tid >= B_ * FCD) return;
  int b = tid / FCD, f = tid % FCD;
  float s = b0[f];
  for (int k = 0; k < CD; ++k) s += gv[b * CD + k] * W0[(size_t)k * FCD + f];
  for (int k = 0; k < SD; ++k)
    s += sent[b * SD + k] * W0[(size_t)(CD + k) * FCD + f];
  h1[tid] = s;
}

__global__ void k_head(const float* __restrict__ h1, const float* __restrict__ g,
                       const float* __restrict__ bta, const float* __restrict__ W1,
                       const float* __restrict__ b1, float* __restrict__ logits) {
  int b = blockIdx.x * blockDim.x + threadIdx.x;
  if (b >= B_) return;
  const float* r = h1 + (size_t)b * FCD;
  float mu = 0.f;
  for (int f = 0; f < FCD; ++f) mu += r[f];
  mu /= (float)FCD;
  float var = 0.f;
  for (int f = 0; f < FCD; ++f) { float d = r[f] - mu; var += d * d; }
  var /= (float)FCD;
  float inv = rsqrtf(var + 1e-5f);
  float s = b1[0];
  for (int f = 0; f < FCD; ++f)
    s += gelu_erf((r[f] - mu) * inv * g[f] + bta[f]) * W1[f];
  logits[b] = s;
}

// ---------------------------------------------------------------------------
extern "C" void kernel_launch(void* const* d_in, const int* in_sizes, int n_in,
                              void* d_out, int out_size, void* d_ws, size_t ws_size,
                              hipStream_t stream) {
  const float* sent     = (const float*)d_in[0];
  const int*   concepts = (const int*)  d_in[1];
  const float* adj      = (const float*)d_in[2];
  const int*   adj_len  = (const int*)  d_in[3];
  const float* emb      = (const float*)d_in[4];
  const float* cptW     = (const float*)d_in[5];
  const float* cptb     = (const float*)d_in[6];
  const float* gw       = (const float*)d_in[7];
  const float* wq       = (const float*)d_in[8];
  const float* bq       = (const float*)d_in[9];
  const float* wk       = (const float*)d_in[10];
  const float* bk       = (const float*)d_in[11];
  const float* wv       = (const float*)d_in[12];
  const float* bv       = (const float*)d_in[13];
  const float* W0       = (const float*)d_in[14];
  const float* b0       = (const float*)d_in[15];
  const float* lng      = (const float*)d_in[16];
  const float* lnb      = (const float*)d_in[17];
  const float* W1       = (const float*)d_in[18];
  const float* b1       = (const float*)d_in[19];
  (void)in_sizes; (void)n_in; (void)out_size; (void)ws_size;

  char* ws = (char*)d_ws;
  size_t off = 0;
  auto alloc = [&](size_t bytes) -> void* {
    void* p = ws + off;
    off += (bytes + 255) & ~(size_t)255;
    return p;
  };
  float* rdeg  = (float*)alloc(sizeof(float) * B_ * NREL * NN);
  bf16*  xbA   = (bf16*) alloc(sizeof(bf16)  * B_ * NP * NP);
  bf16*  xbB   = (bf16*) alloc(sizeof(bf16)  * B_ * NP * NP);
  bf16*  wb    = (bf16*) alloc(sizeof(bf16)  * (size_t)NP * WCOLS);
  bf16*  hbuf  = (bf16*) alloc(sizeof(bf16)  * (size_t)B_ * NP * WCOLS);
  float* xf    = (float*)alloc(sizeof(float) * (size_t)B_ * NN * CD);
  float* qsb   = (float*)alloc(sizeof(float) * B_ * CD);
  float* ksb   = (float*)alloc(sizeof(float) * (size_t)B_ * NN * CD);
  float* vsb   = (float*)alloc(sizeof(float) * (size_t)B_ * NN * CD);
  float* attnw = (float*)alloc(sizeof(float) * B_ * 2 * NN);
  float* gvb   = (float*)alloc(sizeof(float) * B_ * CD);
  float* h1b   = (float*)alloc(sizeof(float) * B_ * FCD);

  // Zero padded bf16 x buffers (pads must stay 0 across layers).
  hipMemsetAsync(xbA, 0, sizeof(bf16) * B_ * NP * NP, stream);
  hipMemsetAsync(xbB, 0, sizeof(bf16) * B_ * NP * NP, stream);

  k_rdeg<<<(B_ * NREL * NN + 255) / 256, 256, 0, stream>>>(adj, rdeg);

  {
    int tiles = B_ * 13 * 13;
    k_embed_gemm<<<(tiles + 3) / 4, 128, 0, stream>>>(concepts, emb, cptW, cptb, xbA);
  }

  bf16* cur = xbA;
  bf16* nxt = xbB;
  for (int l = 0; l < NLAY; ++l) {
    k_convert_w<<<(NP * WCOLS + 255) / 256, 256, 0, stream>>>(
        gw + (size_t)l * CD * WCOLS, wb);
    int tiles = B_ * (NP / 16) * (WCOLS / 16);
    k_layer_gemm<<<(tiles + 3) / 4, 128, 0, stream>>>(cur, wb, hbuf);
    dim3 gdim(4, 13, B_);
    k_agg_gemm<<<gdim, 128, 0, stream>>>(adj, rdeg, hbuf, nxt, xf);
    bf16* t = cur; cur = nxt; nxt = t;
  }

  float* out = (float*)d_out;           // [0,32) logits ; [32, 32+12800) pool_attn
  k_qs<<<(B_ * CD + 255) / 256, 256, 0, stream>>>(sent, wq, bq, qsb);
  k_kv<<<(B_ * NN * CD + 255) / 256, 256, 0, stream>>>(xf, wk, bk, wv, bv, ksb, vsb);
  k_attn<<<(B_ * 2 * NN + 255) / 256, 256, 0, stream>>>(qsb, ksb, attnw);
  k_softmax<<<1, 64, 0, stream>>>(adj_len, attnw, out + B_);
  k_pool<<<(B_ * CD + 255) / 256, 256, 0, stream>>>(attnw, vsb, gvb);
  k_mlp1<<<(B_ * FCD + 255) / 256, 256, 0, stream>>>(gvb, sent, W0, b0, h1b);
  k_head<<<1, B_, 0, stream>>>(h1b, lng, lnb, W1, b1, out);
}
